// My_QuantModule_77189152244284
// MI455X (gfx1250) — matrix operand order repaired and checked by reading, not compiled, not run on hardware
//
#include <hip/hip_runtime.h>

typedef __attribute__((ext_vector_type(16))) _Float16 v16h;
typedef __attribute__((ext_vector_type(8)))  _Float16 v8h;
typedef __attribute__((ext_vector_type(4)))  _Float16 v4h;
typedef __attribute__((ext_vector_type(2)))  _Float16 v2h;
typedef __attribute__((ext_vector_type(8)))  float    v8f;
typedef __attribute__((ext_vector_type(4)))  int      v4i;

#define MDIM 8192
#define NDIM 4096
#define KDIM 4096

#define BM 128
#define BN 128
#define BK 32
#define LSTR 40                 // halves per LDS row: 80 B, 16B-aligned, 20-bank stride
#define KTILES (KDIM / BK)      // 128

// --------------------------- CDNA5 async-to-LDS ----------------------------
#if __has_builtin(__builtin_amdgcn_global_load_async_to_lds_b128)
#define HAVE_ASYNC_LDS 1
#else
#define HAVE_ASYNC_LDS 0
#endif

#if __has_builtin(__builtin_amdgcn_s_wait_asynccnt)
#define WAIT_ASYNC() __builtin_amdgcn_s_wait_asynccnt(0)
#else
#define WAIT_ASYNC() asm volatile("s_wait_asynccnt 0" ::: "memory")
#endif

#if HAVE_ASYNC_LDS
typedef __attribute__((address_space(1))) v4i glob_v4i;
typedef __attribute__((address_space(3))) v4i lds_v4i;
__device__ __forceinline__ void async_copy16(const _Float16* g, _Float16* l) {
  __builtin_amdgcn_global_load_async_to_lds_b128((glob_v4i*)g, (lds_v4i*)l,
                                                 /*offset=*/0, /*cpol=*/0);
}
#endif

// ---------------------------------------------------------------------------
// Kernel 1: group-wise (v=64) symmetric fake-quant of x, output f16.
// One wave32 per group of 64 along K. rint = round-half-even (matches jnp.round).
// ---------------------------------------------------------------------------
__global__ __launch_bounds__(256) void quant_x_kernel(const float* __restrict__ x,
                                                      _Float16* __restrict__ xq) {
  const int lane = threadIdx.x & 31;
  const int wave = threadIdx.x >> 5;
  const long g    = (long)blockIdx.x * 8 + wave;   // group index
  const long base = g * 64 + lane * 2;

  float2 v = *(const float2*)(x + base);
  float am = fmaxf(fabsf(v.x), fabsf(v.y));
#pragma unroll
  for (int off = 16; off > 0; off >>= 1)
    am = fmaxf(am, __shfl_xor(am, off, 32));

  // delta = max(2*absmax/254, 1e-5) = max(absmax/127, 1e-5)
  const float delta = fmaxf(am * (1.0f / 127.0f), 1e-5f);
  const float inv   = 1.0f / delta;
  const float q0 = fminf(fmaxf(__builtin_rintf(v.x * inv), -127.0f), 127.0f) * delta;
  const float q1 = fminf(fmaxf(__builtin_rintf(v.y * inv), -127.0f), 127.0f) * delta;

  v2h h;
  h.x = (_Float16)q0;
  h.y = (_Float16)q1;
  *(v2h*)(xq + base) = h;
}

// ---------------------------------------------------------------------------
// Kernel 2: W f32 -> f16 (bandwidth-bound stream conversion).
// ---------------------------------------------------------------------------
__global__ __launch_bounds__(256) void conv_w_kernel(const float* __restrict__ W,
                                                     _Float16* __restrict__ Wh) {
  const long i = ((long)blockIdx.x * 256 + threadIdx.x) * 4;
  float4 v = *(const float4*)(W + i);
  v4h h;
  h.x = (_Float16)v.x;
  h.y = (_Float16)v.y;
  h.z = (_Float16)v.z;
  h.w = (_Float16)v.w;
  *(v4h*)(Wh + i) = h;
}

// ---------------------------------------------------------------------------
// Fragment loaders from LDS, matching CDNA5 WMMA 16-bit operand layouts.
// A (16x32, MxK): lane L -> row L%16; halves j<8 : K = j + 8*(L/16)
//                                    j>=8: K = 16 + (j-8) + 8*(L/16)
// B (32x16, KxN): lane L -> col L%16; same K mapping as A
// ---------------------------------------------------------------------------
__device__ __forceinline__ v16h frag_a(const _Float16* s, int row, int half) {
  const v8h* p0 = (const v8h*)(s + row * LSTR + half * 8);
  const v8h* p1 = (const v8h*)(s + row * LSTR + 16 + half * 8);
  v8h lo = *p0, hi = *p1;
  return __builtin_shufflevector(lo, hi, 0, 1, 2, 3, 4, 5, 6, 7,
                                 8, 9, 10, 11, 12, 13, 14, 15);
}

__device__ __forceinline__ v16h frag_b(const _Float16* s, int row, int half) {
  const v8h* p0 = (const v8h*)(s + row * LSTR + half * 8);
  const v8h* p1 = (const v8h*)(s + row * LSTR + 16 + half * 8);
  v8h lo = *p0, hi = *p1;
  return __builtin_shufflevector(lo, hi, 0, 1, 2, 3, 4, 5, 6, 7,
                                 8, 9, 10, 11, 12, 13, 14, 15);
}

// ---------------------------------------------------------------------------
// Kernel 3: C = A(8192x4096,f16) * W^T(4096x4096,f16) + bias, f32 accum/out.
// NT layout, 128x128 block tile, 8 waves, 32x64 wave tile, double-buffered
// LDS fed by GLOBAL_LOAD_ASYNC_TO_LDS_B128 (fallback: VGPR staging).
// ---------------------------------------------------------------------------
__global__ __launch_bounds__(256) void gemm_wmma_kernel(const _Float16* __restrict__ A,
                                                        const _Float16* __restrict__ Bm,
                                                        const float* __restrict__ bias,
                                                        float* __restrict__ C) {
  __shared__ __align__(16) _Float16 sA[2][BM * LSTR];
  __shared__ __align__(16) _Float16 sB[2][BN * LSTR];

  const int tid    = threadIdx.x;
  const int lane   = tid & 31;
  const int wave   = tid >> 5;
  const int lane16 = lane & 15;
  const int half   = lane >> 4;
  const int waveM  = wave & 3;   // 4 waves along M -> 32 rows each
  const int waveN  = wave >> 2;  // 2 waves along N -> 64 cols each
  const int blockM = blockIdx.y * BM;
  const int blockN = blockIdx.x * BN;

  // Global->LDS staging: 512 16B chunks per tile, 2 per thread per matrix.
  const int r0  = tid >> 2;            // rows r0 and r0+64
  const int kc0 = (tid & 3) * 8;       // K offset in halves within BK

  const _Float16* aBase = A  + (long)blockM * KDIM;
  const _Float16* bBase = Bm + (long)blockN * KDIM;

  const int ldsA0 = r0 * LSTR + kc0;
  const int ldsA1 = (r0 + 64) * LSTR + kc0;

#if HAVE_ASYNC_LDS
  // ---- prologue: async-fill buffer 0 ----
  async_copy16(aBase + (long)r0 * KDIM + kc0,        &sA[0][ldsA0]);
  async_copy16(aBase + (long)(r0 + 64) * KDIM + kc0, &sA[0][ldsA1]);
  async_copy16(bBase + (long)r0 * KDIM + kc0,        &sB[0][ldsA0]);
  async_copy16(bBase + (long)(r0 + 64) * KDIM + kc0, &sB[0][ldsA1]);
  WAIT_ASYNC();
  __syncthreads();
#else
  uint4 ra0, ra1, rb0, rb1;
  ra0 = *(const uint4*)(aBase + (long)r0 * KDIM + kc0);
  ra1 = *(const uint4*)(aBase + (long)(r0 + 64) * KDIM + kc0);
  rb0 = *(const uint4*)(bBase + (long)r0 * KDIM + kc0);
  rb1 = *(const uint4*)(bBase + (long)(r0 + 64) * KDIM + kc0);
  *(uint4*)(&sA[0][ldsA0]) = ra0;
  *(uint4*)(&sA[0][ldsA1]) = ra1;
  *(uint4*)(&sB[0][ldsA0]) = rb0;
  *(uint4*)(&sB[0][ldsA1]) = rb1;
  __syncthreads();
#endif

  v8f acc[2][4];
#pragma unroll
  for (int i = 0; i < 2; ++i)
#pragma unroll
    for (int j = 0; j < 4; ++j)
      acc[i][j] = (v8f)(0.0f);

  for (int kt = 0; kt < KTILES; ++kt) {
    const int cur = kt & 1;
    const int nxt = cur ^ 1;

    // Issue next tile's loads before compute so they overlap the WMMAs.
    if (kt + 1 < KTILES) {
      const long k = (long)(kt + 1) * BK;
#if HAVE_ASYNC_LDS
      async_copy16(aBase + (long)r0 * KDIM + k + kc0,        &sA[nxt][ldsA0]);
      async_copy16(aBase + (long)(r0 + 64) * KDIM + k + kc0, &sA[nxt][ldsA1]);
      async_copy16(bBase + (long)r0 * KDIM + k + kc0,        &sB[nxt][ldsA0]);
      async_copy16(bBase + (long)(r0 + 64) * KDIM + k + kc0, &sB[nxt][ldsA1]);
#else
      ra0 = *(const uint4*)(aBase + (long)r0 * KDIM + k + kc0);
      ra1 = *(const uint4*)(aBase + (long)(r0 + 64) * KDIM + k + kc0);
      rb0 = *(const uint4*)(bBase + (long)r0 * KDIM + k + kc0);
      rb1 = *(const uint4*)(bBase + (long)(r0 + 64) * KDIM + k + kc0);
#endif
    }
    if (kt + 2 < KTILES) {
      const long k2 = (long)(kt + 2) * BK;
      __builtin_prefetch(aBase + (long)r0 * KDIM + k2 + kc0, 0, 1);
      __builtin_prefetch(bBase + (long)r0 * KDIM + k2 + kc0, 0, 1);
    }

    v16h af[2], bf[4];
#pragma unroll
    for (int i = 0; i < 2; ++i)
      af[i] = frag_a(sA[cur], waveM * 32 + i * 16 + lane16, half);
#pragma unroll
    for (int j = 0; j < 4; ++j)
      bf[j] = frag_b(sB[cur], waveN * 64 + j * 16 + lane16, half);

#pragma unroll
    for (int i = 0; i < 2; ++i)
#pragma unroll
      for (int j = 0; j < 4; ++j)
        acc[i][j] = __builtin_amdgcn_wmma_f32_16x16x32_f16(
            false, af[i], false, bf[j], (short)0, acc[i][j], false, false);

#if HAVE_ASYNC_LDS
    WAIT_ASYNC();              // own async writes into buf[nxt] complete
#else
    if (kt + 1 < KTILES) {
      *(uint4*)(&sA[nxt][ldsA0]) = ra0;
      *(uint4*)(&sA[nxt][ldsA1]) = ra1;
      *(uint4*)(&sB[nxt][ldsA0]) = rb0;
      *(uint4*)(&sB[nxt][ldsA1]) = rb1;
    }
#endif
    __syncthreads();           // all waves' fills of buf[nxt] visible
  }

  // Epilogue: C layout -> lane = N, VGPR v = M%8, half-wave = M/8. Add bias.
#pragma unroll
  for (int j = 0; j < 4; ++j) {
    const int col = blockN + waveN * 64 + j * 16 + lane16;
    const float bv = bias[col];
#pragma unroll
    for (int i = 0; i < 2; ++i) {
      const int rowBase = blockM + waveM * 32 + i * 16 + half * 8;
#pragma unroll
      for (int v = 0; v < 8; ++v)
        C[(long)(rowBase + v) * NDIM + col] = acc[i][j][v] + bv;
    }
  }
}

// ---------------------------------------------------------------------------
extern "C" void kernel_launch(void* const* d_in, const int* in_sizes, int n_in,
                              void* d_out, int out_size, void* d_ws, size_t ws_size,
                              hipStream_t stream) {
  (void)in_sizes; (void)n_in; (void)out_size; (void)ws_size;

  const float* x    = (const float*)d_in[0];   // 8192 x 4096
  const float* W    = (const float*)d_in[1];   // 4096 x 4096
  const float* bias = (const float*)d_in[2];   // 4096
  float* out        = (float*)d_out;           // 8192 x 4096

  _Float16* xq = (_Float16*)d_ws;                          // 64 MiB
  _Float16* Wh = xq + (size_t)MDIM * KDIM;                 // +32 MiB

  // 1) fake-quant x -> f16  (groups: 8192*4096/64 = 524288; 8 groups/block)
  quant_x_kernel<<<(MDIM * (KDIM / 64)) / 8, 256, 0, stream>>>(x, xq);

  // 2) W f32 -> f16  (16777216 elems / 4 per thread / 256 per block)
  conv_w_kernel<<<((size_t)KDIM * NDIM) / 4 / 256, 256, 0, stream>>>(W, Wh);

  // 3) WMMA GEMM + bias
  dim3 grid(NDIM / BN, MDIM / BM);   // (32, 64)
  gemm_wmma_kernel<<<grid, 256, 0, stream>>>(xq, Wh, bias, out);
}
